// Seq2SeqWithAttention_71451075936928
// MI455X (gfx1250) — matrix operand order repaired and checked by reading, not compiled, not run on hardware
//
#include <hip/hip_runtime.h>
#include <hip/hip_bf16.h>

typedef __attribute__((ext_vector_type(16))) _Float16 v16h;
typedef __attribute__((ext_vector_type(8)))  float    v8f;

#define DEVFN __device__ __forceinline__

// Model dims
#define BB 64
#define TF 64
#define TE 32
#define EE 40
#define HH 100
#define H3 300
#define H2 200
#define SPD 256
#define EVN 20000

// ---------------- WMMA helpers (gfx1250, wave32) ----------------
// All operands are stored PRE-SWIZZLED in fragment order:
//   element index = ((tile*32 + lane)*16 + e)
// so each lane loads 16 contiguous halves (32B) -> 2x b128 loads.
DEVFN v16h load_frag(const _Float16* base, int tile) {
    int lane = threadIdx.x & 31;
    return *(const v16h*)(base + ((size_t)tile * 32 + lane) * 16);
}

DEVFN v8f wmma_f16(v16h a, v16h b, v8f c) {
    return __builtin_amdgcn_wmma_f32_16x16x32_f16(
        false, a, false, b, (short)0, c, false, false);
}

// A-fragment K offset within a 32-wide K tile for (lane, e).
// ISA 7.12.2 16-bit A 16x32: k = (q&3)*2 + (q>>2)*16 + (lane>=16)*8 + (e&1), q=e>>1
DEVFN int a_k_of(int lane, int e) {
    int half = lane >> 4;
    int q = e >> 1;
    return ((q & 3) * 2) + ((q >> 2) * 16) + half * 8 + (e & 1);
}

// Inverse map: packed-A linear index for logical element (m, k), Ktiles in K dim.
DEVFN int a_packed_idx(int m, int k, int Ktiles) {
    int mt = m >> 4, kt = k >> 5, kk = k & 31;
    int half = (kk >> 3) & 1;
    int lsb  = kk & 1;
    int qlo  = (kk >> 1) & 3;
    int qhi  = (kk >> 4) & 1;
    int e    = (((qhi << 2) | qlo) << 1) | lsb;
    int lane = (half << 4) | (m & 15);
    return ((mt * Ktiles + kt) * 32 + lane) * 16 + e;
}

DEVFN float sigmoidf_(float x) { return 1.f / (1.f + expf(-x)); }

// ------------- pack f32 B matrix [Ktot x Ntot] -> fragment-ordered f16 -------------
// B 32x16 layout: N=n0+(lane&15), K=k0+(e>>3)*16+(lane>>4)*8+(e&7)
__global__ void k_pack_b(const float* __restrict__ W, _Float16* __restrict__ dst,
                         int Ktot, int Ntot, int KT, int NT) {
    int i = blockIdx.x * 256 + threadIdx.x;
    if (i >= KT * NT * 512) return;
    int e = i & 15;
    int lane = (i >> 4) & 31;
    int tile = i >> 9;
    int nt = tile % NT, kt = tile / NT;
    int n = nt * 16 + (lane & 15);
    int k = kt * 32 + (e >> 3) * 16 + (lane >> 4) * 8 + (e & 7);
    dst[i] = (k < Ktot && n < Ntot) ? (_Float16)W[k * Ntot + n] : (_Float16)0.f;
}

// ---------------- embed + xs = emb[tok] @ enc_W + enc_b[0] ----------------
__global__ void k_embed_xs(const int* __restrict__ tok,
                           const float* __restrict__ emb,
                           const float* __restrict__ W,
                           const float* __restrict__ b0,
                           float* __restrict__ xs) {
    int i = blockIdx.x * 256 + threadIdx.x;
    if (i >= BB * TF * H3) return;
    int row = i / H3, j = i % H3;
    int t = tok[row];
    const float* er = emb + (size_t)t * EE;
    float acc = b0[j];
    for (int e = 0; e < EE; ++e) acc += er[e] * W[e * H3 + j];
    xs[i] = acc;
}

// ---------------- encoder GRU: single persistent workgroup (f32 VALU) ----------------
__global__ void k_enc_gru(const float* __restrict__ xs,
                          const float* __restrict__ U,
                          const float* __restrict__ bvec,
                          float* __restrict__ enc_out,
                          float* __restrict__ h_state) {
    extern __shared__ unsigned char dynsm[];
    float* h  = (float*)dynsm;
    float* hs = h + BB * HH;
    int tid = threadIdx.x;
    const int NT = 1024;
    for (int i = tid; i < BB * HH; i += NT) h[i] = 0.f;
    __syncthreads();
    for (int t = 0; t < TF; ++t) {
        for (int i = tid; i < BB * H3; i += NT) {
            int b = i / H3, j = i % H3;
            const float* hb = &h[b * HH];
            float acc = bvec[H3 + j];
            for (int k = 0; k < HH; ++k) acc += hb[k] * U[k * H3 + j];
            hs[i] = acc;
        }
        __syncthreads();
        for (int i = tid; i < BB * HH; i += NT) {
            int b = i / HH, j = i % HH;
            const float* xr = xs + ((size_t)b * TF + t) * H3;
            float z = sigmoidf_(xr[j] + hs[b * H3 + j]);
            float r = sigmoidf_(xr[HH + j] + hs[b * H3 + HH + j]);
            float cand = tanhf(xr[2 * HH + j] + r * hs[b * H3 + 2 * HH + j]);
            float hn = z * h[i] + (1.f - z) * cand;
            h[i] = hn;
            enc_out[((size_t)b * TF + t) * HH + j] = hn;
        }
        __syncthreads();
    }
    for (int i = tid; i < BB * HH; i += NT) h_state[i] = h[i];
}

// ---------------- decoder attention (one block per batch b) ----------------
// A: 64 x 224 (concat(h,enc), K-pad) staged in LDS in fragment order.
#define ATT_KT 7
__global__ void k_dec_attn(const float* __restrict__ h_state,
                           const float* __restrict__ enc,
                           const _Float16* __restrict__ W1p,  // packed [7 x 13] tiles
                           const float* __restrict__ W2,
                           float* __restrict__ readbuf) {
    __shared__ __align__(32) _Float16 Ap[4 * ATT_KT * 512];   // 14336 halves
    __shared__ float hloc[HH];
    __shared__ float qsh[208];
    __shared__ float sc[TF];
    int b = blockIdx.x, tid = threadIdx.x;
    for (int i = tid; i < HH; i += 256) hloc[i] = h_state[b * HH + i];
    for (int i = tid; i < TF; i += 256) sc[i] = 0.f;
    __syncthreads();
    for (int i = tid; i < 4 * ATT_KT * 512; i += 256) {
        int e = i & 15, lane = (i >> 4) & 31, tile = i >> 9;
        int kt = tile % ATT_KT, mt = tile / ATT_KT;
        int m = mt * 16 + (lane & 15);
        int k = kt * 32 + a_k_of(lane, e);
        float v = (k < HH) ? hloc[k]
                : (k < H2) ? enc[((size_t)b * TF + m) * HH + (k - HH)]
                : 0.f;
        Ap[i] = (_Float16)v;
    }
    for (int n = tid; n < 208; n += 256) {
        float acc = 0.f;
        if (n < H2) { for (int k = 0; k < HH; ++k) acc += hloc[k] * W2[k * H2 + n]; }
        qsh[n] = acc;
    }
    __syncthreads();

    int wave = tid >> 5, lane = tid & 31, half = lane >> 4;
    for (int tile = wave; tile < 4 * 13; tile += 8) {
        int mt = tile / 13, nt = tile % 13;
        v8f c = {};
        for (int kc = 0; kc < ATT_KT; ++kc)
            c = wmma_f16(load_frag(Ap, mt * ATT_KT + kc),
                         load_frag(W1p, kc * 13 + nt), c);
        int n = nt * 16 + (lane & 15);
        float qv = qsh[n];
#pragma unroll
        for (int v = 0; v < 8; ++v)
            atomicAdd(&sc[mt * 16 + v + 8 * half], tanhf(c[v]) * qv);
    }
    __syncthreads();
    if (tid == 0) {
        float mx = -3.4e38f;
        for (int t = 0; t < TF; ++t) mx = fmaxf(mx, sc[t]);
        float s = 0.f;
        for (int t = 0; t < TF; ++t) { float e = expf(sc[t] - mx); sc[t] = e; s += e; }
        float inv = 1.f / s;
        for (int t = 0; t < TF; ++t) sc[t] *= inv;
    }
    __syncthreads();
    for (int n = tid; n < HH; n += 256) {
        float acc = 0.f;
        for (int t = 0; t < TF; ++t) acc += sc[t] * enc[((size_t)b * TF + t) * HH + n];
        readbuf[b * HH + n] = acc;
    }
}

// ---------------- decoder GRU (tiny, single workgroup, f32) ----------------
__global__ void k_dec_gru(float* __restrict__ h_state,
                          const float* __restrict__ readbuf,
                          const float* __restrict__ W,
                          const float* __restrict__ U,
                          const float* __restrict__ bvec,
                          float* __restrict__ dec_out, int step) {
    __shared__ float h[BB * HH];
    __shared__ float rd[BB * HH];
    int tid = threadIdx.x;
    const int NT = 1024;
    for (int i = tid; i < BB * HH; i += NT) { h[i] = h_state[i]; rd[i] = readbuf[i]; }
    __syncthreads();
    float hn[7];
    int cnt = 0;
    for (int i = tid; i < BB * HH; i += NT) {
        int b = i / HH, j = i % HH;
        float xz = bvec[j], xr = bvec[HH + j], xh = bvec[2 * HH + j];
        float hz = bvec[H3 + j], hr = bvec[H3 + HH + j], hh = bvec[H3 + 2 * HH + j];
        const float* rb = &rd[b * HH];
        const float* hb = &h[b * HH];
        for (int k = 0; k < HH; ++k) {
            float rk = rb[k], hk = hb[k];
            xz += rk * W[k * H3 + j];
            xr += rk * W[k * H3 + HH + j];
            xh += rk * W[k * H3 + 2 * HH + j];
            hz += hk * U[k * H3 + j];
            hr += hk * U[k * H3 + HH + j];
            hh += hk * U[k * H3 + 2 * HH + j];
        }
        float z = sigmoidf_(xz + hz);
        float r = sigmoidf_(xr + hr);
        float cand = tanhf(xh + r * hh);
        hn[cnt++] = z * h[i] + (1.f - z) * cand;
    }
    cnt = 0;
    for (int i = tid; i < BB * HH; i += NT) {
        int b = i / HH, j = i % HH;
        h_state[i] = hn[cnt];
        dec_out[((size_t)b * TE + step) * HH + j] = hn[cnt];
        ++cnt;
    }
}

// ------------- scratchpad: relu(z@sp1)@sp2 -> mean -> gate, update enc -------
// A1: 64 x 320 (concat(h,read,enc)) packed; GEMM1 output scattered straight
// into packed-A layout for GEMM2 (64 x 256).
#define SC_KT1 10
#define SC_KT2 8
__global__ void k_dec_scratch(const float* __restrict__ h_state,
                              const float* __restrict__ readbuf,
                              float* __restrict__ enc,
                              const _Float16* __restrict__ sp1p, // packed [10 x 16]
                              const float* __restrict__ sp1b,
                              const _Float16* __restrict__ sp2p, // packed [8 x 16]
                              const float* __restrict__ sp2b) {
    extern __shared__ unsigned char dynsm[];
    _Float16* A1 = (_Float16*)dynsm;                             // 4*10*512
    _Float16* R2 = (_Float16*)(dynsm + 4 * SC_KT1 * 512 * 2);    // 4*8*512
    float* rowsum = (float*)(dynsm + (4 * SC_KT1 + 4 * SC_KT2) * 512 * 2);
    float* hn = rowsum + BB;
    float* rd = hn + HH;
    int b = blockIdx.x, tid = threadIdx.x;
    for (int i = tid; i < HH; i += 256) { hn[i] = h_state[b * HH + i]; rd[i] = readbuf[b * HH + i]; }
    for (int i = tid; i < BB; i += 256) rowsum[i] = 0.f;
    __syncthreads();
    for (int i = tid; i < 4 * SC_KT1 * 512; i += 256) {
        int e = i & 15, lane = (i >> 4) & 31, tile = i >> 9;
        int kt = tile % SC_KT1, mt = tile / SC_KT1;
        int m = mt * 16 + (lane & 15);
        int k = kt * 32 + a_k_of(lane, e);
        float v = (k < HH) ? hn[k]
                : (k < H2) ? rd[k - HH]
                : (k < H3) ? enc[((size_t)b * TF + m) * HH + (k - H2)]
                : 0.f;
        A1[i] = (_Float16)v;
    }
    __syncthreads();

    int wave = tid >> 5, lane = tid & 31, half = lane >> 4;
    // GEMM1: [64,300+] x [300,256] + bias, relu -> R2 (packed-A order)
    for (int tile = wave; tile < 4 * 16; tile += 8) {
        int mt = tile >> 4, nt = tile & 15;
        int n = nt * 16 + (lane & 15);
        v8f c = {};
        float bv = sp1b[n];
#pragma unroll
        for (int v = 0; v < 8; ++v) c[v] = bv;
        for (int kc = 0; kc < SC_KT1; ++kc)
            c = wmma_f16(load_frag(A1, mt * SC_KT1 + kc),
                         load_frag(sp1p, kc * 16 + nt), c);
#pragma unroll
        for (int v = 0; v < 8; ++v) {
            int m = mt * 16 + v + 8 * half;
            R2[a_packed_idx(m, n, SC_KT2)] = (_Float16)fmaxf(c[v], 0.f);
        }
    }
    __syncthreads();
    // GEMM2: [64,256] x [256,256] + bias -> row sums (mean)
    for (int tile = wave; tile < 4 * 16; tile += 8) {
        int mt = tile >> 4, nt = tile & 15;
        int n = nt * 16 + (lane & 15);
        v8f c = {};
        float bv = sp2b[n];
#pragma unroll
        for (int v = 0; v < 8; ++v) c[v] = bv;
        for (int kc = 0; kc < SC_KT2; ++kc)
            c = wmma_f16(load_frag(R2, mt * SC_KT2 + kc),
                         load_frag(sp2p, kc * 16 + nt), c);
#pragma unroll
        for (int v = 0; v < 8; ++v)
            atomicAdd(&rowsum[mt * 16 + v + 8 * half], c[v]);
    }
    __syncthreads();
    for (int i = tid; i < TF * HH; i += 256) {
        int t = i / HH, n = i % HH;
        float g = sigmoidf_(rowsum[t] * (1.f / SPD));
        size_t idx = ((size_t)b * TF + t) * HH + n;
        enc[idx] = g * enc[idx] + (1.f - g) * rd[n];
    }
}

// ---------------- head ----------------
// ff1: relu(dec_out@ff1_W+b) written DIRECTLY in packed-A fragment order (2048 x 128pad)
__global__ void k_ff1_pack(const float* __restrict__ dec_out,
                           const float* __restrict__ W,
                           const float* __restrict__ bias,
                           _Float16* __restrict__ hidp) {  // 128 mtiles * 4 ktiles * 512
    int i = blockIdx.x * 256 + threadIdx.x;
    if (i >= 128 * 4 * 512) return;
    int e = i & 15, lane = (i >> 4) & 31, tile = i >> 9;
    int kt = tile & 3, mt = tile >> 2;
    int m = mt * 16 + (lane & 15);
    int k = kt * 32 + a_k_of(lane, e);
    float v = 0.f;
    if (k < HH) {
        v = bias[k];
        const float* dr = dec_out + (size_t)m * HH;
        for (int kk = 0; kk < HH; ++kk) v += dr[kk] * W[kk * HH + k];
        v = fmaxf(v, 0.f);
    }
    hidp[i] = (_Float16)v;
}

// ff2: [2048,100] x [100,20000] + bias. N = 1250 ntiles exactly -> no edge guards.
__global__ void k_ff2(const _Float16* __restrict__ hidp,
                      const _Float16* __restrict__ Wp,   // packed [4 x 1250]
                      const float* __restrict__ bias,
                      float* __restrict__ out) {
    int wave = threadIdx.x >> 5, lane = threadIdx.x & 31, half = lane >> 4;
    int mt  = blockIdx.y * 4 + (wave >> 1);   // 32 * 4 = 128 mtiles
    int ntg = blockIdx.x * 2 + (wave & 1);    // 625 * 2 = 1250 ntiles
    int n = ntg * 16 + (lane & 15);
    v8f c = {};
    float bv = bias[n];
#pragma unroll
    for (int v = 0; v < 8; ++v) c[v] = bv;
    for (int kc = 0; kc < 4; ++kc)
        c = wmma_f16(load_frag(hidp, mt * 4 + kc),
                     load_frag(Wp, kc * 1250 + ntg), c);
#pragma unroll
    for (int v = 0; v < 8; ++v)
        out[(size_t)(mt * 16 + v + 8 * half) * EVN + n] = c[v];
}

__global__ void k_softmax_rows(float* __restrict__ out) {
    __shared__ float red[256];
    int row = blockIdx.x, tid = threadIdx.x;
    float* p = out + (size_t)row * EVN;
    float mx = -3.4e38f;
    for (int i = tid; i < EVN; i += 256) mx = fmaxf(mx, p[i]);
    red[tid] = mx; __syncthreads();
    for (int s = 128; s > 0; s >>= 1) { if (tid < s) red[tid] = fmaxf(red[tid], red[tid + s]); __syncthreads(); }
    mx = red[0]; __syncthreads();
    float sum = 0.f;
    for (int i = tid; i < EVN; i += 256) sum += expf(p[i] - mx);
    red[tid] = sum; __syncthreads();
    for (int s = 128; s > 0; s >>= 1) { if (tid < s) red[tid] += red[tid + s]; __syncthreads(); }
    float inv = 1.f / red[0];
    for (int i = tid; i < EVN; i += 256) p[i] = expf(p[i] - mx) * inv;
}

// ---------------- host launcher ----------------
extern "C" void kernel_launch(void* const* d_in, const int* in_sizes, int n_in,
                              void* d_out, int out_size, void* d_ws, size_t ws_size,
                              hipStream_t stream) {
    const int*   enc_in = (const int*)d_in[0];
    const float* fr_emb = (const float*)d_in[2];
    const float* enc_W  = (const float*)d_in[4];
    const float* enc_U  = (const float*)d_in[5];
    const float* enc_b  = (const float*)d_in[6];
    const float* dec_W  = (const float*)d_in[7];
    const float* dec_U  = (const float*)d_in[8];
    const float* dec_b  = (const float*)d_in[9];
    const float* att_W1 = (const float*)d_in[10];
    const float* att_W2 = (const float*)d_in[11];
    const float* sp1_W  = (const float*)d_in[12];
    const float* sp1_b  = (const float*)d_in[13];
    const float* sp2_W  = (const float*)d_in[14];
    const float* sp2_b  = (const float*)d_in[15];
    const float* ff1_W  = (const float*)d_in[16];
    const float* ff1_b  = (const float*)d_in[17];
    const float* ff2_W  = (const float*)d_in[18];
    const float* ff2_b  = (const float*)d_in[19];
    float* out = (float*)d_out;

    char* ws = (char*)d_ws;
    size_t off = 0;
    auto take = [&](size_t bytes) { size_t r = off; off = (off + bytes + 255) & ~(size_t)255; return r; };
    float*    xs      = (float*)(ws + take((size_t)BB * TF * H3 * 4));
    float*    enc_out = (float*)(ws + take((size_t)BB * TF * HH * 4));
    float*    enc     = (float*)(ws + take((size_t)BB * TF * HH * 4));
    float*    h_state = (float*)(ws + take((size_t)BB * HH * 4));
    float*    readbuf = (float*)(ws + take((size_t)BB * HH * 4));
    float*    dec_out = (float*)(ws + take((size_t)BB * TE * HH * 4));
    _Float16* hidp    = (_Float16*)(ws + take((size_t)128 * 4 * 512 * 2));
    _Float16* W1p     = (_Float16*)(ws + take((size_t)7 * 13 * 512 * 2));
    _Float16* sp1p    = (_Float16*)(ws + take((size_t)10 * 16 * 512 * 2));
    _Float16* sp2p    = (_Float16*)(ws + take((size_t)8 * 16 * 512 * 2));
    _Float16* ff2p    = (_Float16*)(ws + take((size_t)4 * 1250 * 512 * 2));
    (void)ws_size; (void)in_sizes; (void)n_in; (void)out_size;

    // pack weights into fragment-ordered f16 (one-time, L2-resident thereafter)
    k_pack_b<<<(7 * 13 * 512 + 255) / 256, 256, 0, stream>>>(att_W1, W1p, H2, H2, 7, 13);
    k_pack_b<<<(10 * 16 * 512 + 255) / 256, 256, 0, stream>>>(sp1_W, sp1p, H3, SPD, 10, 16);
    k_pack_b<<<(8 * 16 * 512 + 255) / 256, 256, 0, stream>>>(sp2_W, sp2p, SPD, SPD, 8, 16);
    k_pack_b<<<(4 * 1250 * 512 + 255) / 256, 256, 0, stream>>>(ff2_W, ff2p, HH, EVN, 4, 1250);

    // encoder
    k_embed_xs<<<(BB * TF * H3 + 255) / 256, 256, 0, stream>>>(enc_in, fr_emb, enc_W, enc_b, xs);
    k_enc_gru<<<1, 1024, (BB * HH + BB * H3) * 4, stream>>>(xs, enc_U, enc_b, enc_out, h_state);
    hipMemcpyAsync(enc, enc_out, (size_t)BB * TF * HH * 4, hipMemcpyDeviceToDevice, stream);

    // decoder: 32 sequential steps
    size_t scratch_lds = (size_t)(4 * SC_KT1 + 4 * SC_KT2) * 512 * 2 + (BB + 2 * HH) * 4;
    for (int s = 0; s < TE; ++s) {
        k_dec_attn<<<BB, 256, 0, stream>>>(h_state, enc, W1p, att_W2, readbuf);
        k_dec_gru<<<1, 1024, 0, stream>>>(h_state, readbuf, dec_W, dec_U, dec_b, dec_out, s);
        k_dec_scratch<<<BB, 256, scratch_lds, stream>>>(h_state, readbuf, enc, sp1p, sp1_b, sp2p, sp2_b);
    }

    // head
    k_ff1_pack<<<(128 * 4 * 512 + 255) / 256, 256, 0, stream>>>(dec_out, ff1_W, ff1_b, hidp);
    dim3 g2(625, 32);
    k_ff2<<<g2, 256, 0, stream>>>(hidp, ff2p, ff2_b, out);
    k_softmax_rows<<<BB * TE, 256, 0, stream>>>(out);
}
